// MultiHeadAttention_39350490366252
// MI455X (gfx1250) — compile-verified, matrix operand order checked by reading, not run on hardware
//
#include <hip/hip_runtime.h>
#include <hip/hip_bf16.h>
#include <stdint.h>

// ---------------------------------------------------------------------------
// Types for WMMA fragments (CDNA5 gfx1250, wave32)
// ---------------------------------------------------------------------------
typedef __attribute__((ext_vector_type(16))) __bf16     v16bf;
typedef __attribute__((ext_vector_type(8)))  float      v8f;
typedef __attribute__((ext_vector_type(4)))  unsigned int u32x4;

union Frag16 {            // 32 bytes = 16 bf16 = one WMMA A/B operand
    u32x4 q[2];
    v16bf bf;
};

__device__ __forceinline__ unsigned short f2bf(float f) {
    unsigned int u = __float_as_uint(f);
    unsigned int r = (u + 0x7FFFu + ((u >> 16) & 1u)) >> 16;   // RNE
    return (unsigned short)r;
}

__device__ __forceinline__ v8f wmma_bf16(const Frag16& a, const Frag16& b, v8f c) {
    return __builtin_amdgcn_wmma_f32_16x16x32_bf16(
        /*neg_a=*/false, a.bf, /*neg_b=*/false, b.bf,
        /*c_mod=*/(short)0, c, /*reuse_a=*/false, /*reuse_b=*/false);
}

// ---------------------------------------------------------------------------
// gfx1250 async global->LDS copy (ASYNCcnt path, codegen-verified round 3:
// emits global_load_async_to_lds_b128 / s_wait_asynccnt).
// ---------------------------------------------------------------------------
#define ASYNC_LDS_OK __has_builtin(__builtin_amdgcn_global_load_async_to_lds_b128)

typedef int i32x4g __attribute__((vector_size(16)));
typedef __attribute__((address_space(1))) i32x4g* gas_i32x4_ptr;
typedef __attribute__((address_space(3))) i32x4g* las_i32x4_ptr;

__device__ __forceinline__ void async_copy_b128(unsigned short* lds_dst,
                                                const unsigned short* g_src) {
#if ASYNC_LDS_OK
    __builtin_amdgcn_global_load_async_to_lds_b128(
        (gas_i32x4_ptr)(uintptr_t)g_src,
        (las_i32x4_ptr)(unsigned int)(uintptr_t)lds_dst,
        0, 0);
#else
    *(u32x4*)lds_dst = *(const u32x4*)g_src;
#endif
}

__device__ __forceinline__ void wait_async_all() {
#if __has_builtin(__builtin_amdgcn_s_wait_asynccnt)
    __builtin_amdgcn_s_wait_asynccnt(0);
#elif ASYNC_LDS_OK
    asm volatile("s_wait_asynccnt 0x0" ::: "memory");
#endif
}

// ---------------------------------------------------------------------------
// Problem constants
// ---------------------------------------------------------------------------
#define BATCH 2
#define SEQ   2048
#define DM    1024
#define HEADS 16
#define HDIM  64
#define MTOT  (BATCH * SEQ)      // 4096
#define NTOT  (3 * DM)           // 3072

// ---------------------------------------------------------------------------
// Kernel 1: fp32 -> bf16 conversion (vectorized 4 elems/thread)
// ---------------------------------------------------------------------------
__global__ __launch_bounds__(256) void cvt_f32_to_bf16(
    const float* __restrict__ src, unsigned short* __restrict__ dst, int n)
{
    int idx = (blockIdx.x * 256 + threadIdx.x) * 4;
    if (idx + 3 < n) {
        float4 f = *(const float4*)(src + idx);
        unsigned int lo = (unsigned int)f2bf(f.x) | ((unsigned int)f2bf(f.y) << 16);
        unsigned int hi = (unsigned int)f2bf(f.z) | ((unsigned int)f2bf(f.w) << 16);
        *(uint2*)(dst + idx) = make_uint2(lo, hi);
    }
}

// ---------------------------------------------------------------------------
// Kernel 2: QKV projection GEMM:  qkv[m,f] = sum_c X[m,c] * W[f,c]
// 64x64 tile per block (4 waves x 16 rows). The shared 64x32 W tile is staged
// in LDS via async-to-LDS, double-buffered across K chunks; A fragments come
// from row-major B128 global loads. Outputs scattered into Q/K/V [b][h][t][d].
// ---------------------------------------------------------------------------
__global__ __launch_bounds__(128) void qkv_gemm(
    const unsigned short* __restrict__ X,   // [4096,1024] bf16
    const unsigned short* __restrict__ W,   // [3072,1024] bf16
    unsigned short* __restrict__ Qb,
    unsigned short* __restrict__ Kb,
    unsigned short* __restrict__ Vb)
{
    __shared__ unsigned short lds_b[2][64 * 32];    // [n][k], 4 KB per buffer

    const int tid   = threadIdx.x;
    const int wave  = tid >> 5;
    const int lane  = tid & 31;
    const int lmod  = lane & 15;
    const int lhalf = lane >> 4;

    const int mtile = (blockIdx.x & 63) * 64 + wave * 16;   // 64 m-tiles
    const int ntile = (blockIdx.x >> 6) * 64;               // 48 n-tiles

    // staging assignment: thread -> 32 contiguous bytes of the [64 n][32 k] tile
    const int sn = tid >> 1;                 // 0..63 (n within tile)
    const int sk = (tid & 1) * 16;           // 0 or 16 (k within chunk)
    const unsigned short* wrow = W + (size_t)(ntile + sn) * DM + sk;

    auto stage = [&](int buf, int kc) {
        unsigned short* l = &lds_b[buf][sn * 32 + sk];
        const unsigned short* g = wrow + kc;
        async_copy_b128(l, g);
        async_copy_b128(l + 8, g + 8);
    };

    const unsigned short* arow = X + (size_t)(mtile + lmod) * DM + lhalf * 8;

    v8f acc[4] = {v8f{}, v8f{}, v8f{}, v8f{}};

    stage(0, 0);                                   // prologue fill
    for (int jc = 0; jc < DM / 32; ++jc) {
        const int kc  = jc * 32;
        const int cur = jc & 1;

        wait_async_all();                          // chunk jc resident (this wave)
        __syncthreads();                           // all waves resident + prev reads done
        if (jc + 1 < DM / 32) stage(cur ^ 1, kc + 32);

        if (kc + 64 < DM) __builtin_prefetch(arow + kc + 64, 0, 3);  // near-scope

        Frag16 a;
        a.q[0] = *(const u32x4*)(arow + kc);       // K = kc + lhalf*8 + 0..7
        a.q[1] = *(const u32x4*)(arow + kc + 16);  // K = kc + 16 + lhalf*8 + 0..7
#pragma unroll
        for (int nt = 0; nt < 4; ++nt) {
            const unsigned short* brow = &lds_b[cur][(nt * 16 + lmod) * 32 + lhalf * 8];
            Frag16 b;
            b.q[0] = *(const u32x4*)(brow);
            b.q[1] = *(const u32x4*)(brow + 16);
            acc[nt] = wmma_bf16(a, b, acc[nt]);
        }
    }

    // C/D layout: VGPR r, lanes 0-15 -> M=r, lanes 16-31 -> M=r+8; N = lane%16
#pragma unroll
    for (int nt = 0; nt < 4; ++nt) {
        const int f     = ntile + nt * 16 + lmod;
        const int which = f >> 10;          // 0=Q 1=K 2=V
        const int h     = (f >> 6) & 15;
        const int d     = f & 63;
        unsigned short* dst = (which == 0) ? Qb : (which == 1) ? Kb : Vb;
#pragma unroll
        for (int r = 0; r < 8; ++r) {
            const int m = mtile + r + lhalf * 8;
            const int b = m >> 11;          // /2048
            const int t = m & 2047;
            dst[(((size_t)(b * HEADS + h) * SEQ + t) * HDIM) + d] = f2bf(acc[nt][r]);
        }
    }
}

// ---------------------------------------------------------------------------
// Kernel 3: causal flash attention over bf16 Q/K/V in [b][h][t][d] layout.
// Block = 4 waves = one (b,h) x 64-row Q tile; waves own 16 Q rows each.
// Double-buffered 32-key chunks: K tile async-staged (ASYNCcnt), V tile
// prefetched into VGPRs (B128) and spilled transposed to LDS next iteration.
// One barrier per chunk; fills overlap WMMA + softmax of the previous chunk.
// ---------------------------------------------------------------------------
__global__ __launch_bounds__(128) void attn_fwd(
    const unsigned short* __restrict__ Qb,
    const unsigned short* __restrict__ Kb,
    const unsigned short* __restrict__ Vb,
    float* __restrict__ Out)
{
    __shared__ unsigned short lds_k[2][32 * HDIM];  // [key][d]   2 x 4 KB
    __shared__ unsigned short lds_vt[2][HDIM * 32]; // [d][key]   2 x 4 KB
    __shared__ unsigned short lds_p[4][16 * 32];    // per-wave P 4 KB

    const int tid   = threadIdx.x;
    const int wave  = tid >> 5;
    const int lane  = tid & 31;
    const int lmod  = lane & 15;
    const int lhalf = lane >> 4;

    const int qt = blockIdx.x & 31;                 // q tile index (64 rows)
    const int bh = blockIdx.x >> 5;                 // 0..31 = b*16 + h

    const unsigned short* qbase = Qb + (size_t)bh * SEQ * HDIM;
    const unsigned short* kbase = Kb + (size_t)bh * SEQ * HDIM;
    const unsigned short* vbase = Vb + (size_t)bh * SEQ * HDIM;

    const int qrow0 = qt * 64 + wave * 16;

    // Q A-fragments (d-chunks 0 and 32), resident in VGPRs for whole kernel
    Frag16 qa[2];
    const unsigned short* qrow = qbase + (size_t)(qrow0 + lmod) * HDIM + lhalf * 8;
    qa[0].q[0] = *(const u32x4*)(qrow);
    qa[0].q[1] = *(const u32x4*)(qrow + 16);
    qa[1].q[0] = *(const u32x4*)(qrow + 32);
    qa[1].q[1] = *(const u32x4*)(qrow + 48);

    // staging helpers -------------------------------------------------------
    const int koff = tid * 16;                      // K tile: 32 B per thread
    auto stage_k = [&](int buf, int k0) {
        const unsigned short* g = kbase + (size_t)k0 * HDIM + koff;
        async_copy_b128(&lds_k[buf][koff], g);
        async_copy_b128(&lds_k[buf][koff + 8], g + 8);
    };
    // V tile: 2 x B128 per thread held in VGPRs, spilled transposed later
    auto load_v = [&](int k0, u32x4 vr[2]) {
#pragma unroll
        for (int i = 0; i < 2; ++i) {
            const int e  = (tid + 128 * i) * 8;
            const int kk = e >> 6;
            const int d0 = e & 63;
            vr[i] = *(const u32x4*)(vbase + (size_t)(k0 + kk) * HDIM + d0);
        }
    };
    auto store_v = [&](int buf, const u32x4 vr[2]) {
#pragma unroll
        for (int i = 0; i < 2; ++i) {
            const int e  = (tid + 128 * i) * 8;
            const int kk = e >> 6;
            const int d0 = e & 63;
            const unsigned short* pe = (const unsigned short*)&vr[i];
#pragma unroll
            for (int j = 0; j < 8; ++j)
                lds_vt[buf][(d0 + j) * 32 + kk] = pe[j];
        }
    };

    v8f o[4] = {v8f{}, v8f{}, v8f{}, v8f{}};        // 16 x 64 output accum
    float mrun[8], lrun[8];
#pragma unroll
    for (int r = 0; r < 8; ++r) { mrun[r] = -3.0e38f; lrun[r] = 0.0f; }

    const int nchunk  = (qt + 1) * 2;               // 32-key chunks up to diagonal
    const int qr_lane = qrow0 + lhalf * 8;          // row of accum slot r is qr_lane + r

    u32x4 vreg[2], vnext[2];
    stage_k(0, 0);                                  // prologue: chunk 0 in flight
    load_v(0, vreg);

    for (int jc = 0; jc < nchunk; ++jc) {
        const int k0  = jc * 32;
        const int cur = jc & 1;

        store_v(cur, vreg);                         // spill chunk jc's V tile
        wait_async_all();                           // K chunk jc resident (this wave)
        __syncthreads();                            // all fills visible; prior reads done

        if (jc + 1 < nchunk) {                      // overlap next fills with compute
            stage_k(cur ^ 1, k0 + 32);
            load_v(k0 + 32, vnext);
        }

        // --- scores S[16q x 32k]: 4 WMMAs (2 key sub-tiles x 2 d-chunks) ---
        v8f s0 = {}, s1 = {};
#pragma unroll
        for (int c = 0; c < 2; ++c) {
            Frag16 b0, b1;
            const unsigned short* kr0 = &lds_k[cur][lmod * HDIM + c * 32 + lhalf * 8];
            const unsigned short* kr1 = &lds_k[cur][(16 + lmod) * HDIM + c * 32 + lhalf * 8];
            b0.q[0] = *(const u32x4*)(kr0);
            b0.q[1] = *(const u32x4*)(kr0 + 16);
            b1.q[0] = *(const u32x4*)(kr1);
            b1.q[1] = *(const u32x4*)(kr1 + 16);
            s0 = wmma_bf16(qa[c], b0, s0);
            s1 = wmma_bf16(qa[c], b1, s1);
        }

        // --- scale, causal mask, online softmax (rows live in 16-lane halves) ---
#pragma unroll
        for (int r = 0; r < 8; ++r) {
            const int q = qr_lane + r;
            float a0 = s0[r] * 0.125f;              // 1/sqrt(64)
            float a1 = s1[r] * 0.125f;
            if (k0 + lmod > q)      a0 = -3.0e38f;
            if (k0 + 16 + lmod > q) a1 = -3.0e38f;

            float mx = fmaxf(a0, a1);
#pragma unroll
            for (int m = 1; m <= 8; m <<= 1)
                mx = fmaxf(mx, __shfl_xor(mx, m, 32));

            const float mnew  = fmaxf(mrun[r], mx);
            const float alpha = __expf(mrun[r] - mnew);
            mrun[r] = mnew;

            const float p0 = __expf(a0 - mnew);
            const float p1 = __expf(a1 - mnew);
            float rs = p0 + p1;
#pragma unroll
            for (int m = 1; m <= 8; m <<= 1)
                rs += __shfl_xor(rs, m, 32);
            lrun[r] = lrun[r] * alpha + rs;

            o[0][r] *= alpha; o[1][r] *= alpha; o[2][r] *= alpha; o[3][r] *= alpha;

            const int row = r + lhalf * 8;
            lds_p[wave][row * 32 + lmod]      = f2bf(p0);
            lds_p[wave][row * 32 + 16 + lmod] = f2bf(p1);
        }

        // --- PV: O[16q x 64d] += P[16q x 32k] @ V[32k x 64d] (4 WMMAs) ---
        Frag16 pa;
        const unsigned short* prow = &lds_p[wave][lmod * 32 + lhalf * 8];
        pa.q[0] = *(const u32x4*)(prow);
        pa.q[1] = *(const u32x4*)(prow + 16);
#pragma unroll
        for (int dt = 0; dt < 4; ++dt) {
            Frag16 vb;
            const unsigned short* vtr = &lds_vt[cur][(dt * 16 + lmod) * 32 + lhalf * 8];
            vb.q[0] = *(const u32x4*)(vtr);
            vb.q[1] = *(const u32x4*)(vtr + 16);
            o[dt] = wmma_bf16(pa, vb, o[dt]);
        }

        vreg[0] = vnext[0];
        vreg[1] = vnext[1];
    }

    // --- finalize: O / l, store fp32 to [b][t][h*64+d] ---
    const int b = bh >> 4;
    const int h = bh & 15;
#pragma unroll
    for (int r = 0; r < 8; ++r) {
        const float inv = 1.0f / lrun[r];
        const int   q   = qr_lane + r;
        float* orow = Out + ((size_t)(b * SEQ + q) * DM) + h * HDIM;
#pragma unroll
        for (int dt = 0; dt < 4; ++dt)
            orow[dt * 16 + lmod] = o[dt][r] * inv;
    }
}

// ---------------------------------------------------------------------------
// Host-side launcher
// ---------------------------------------------------------------------------
extern "C" void kernel_launch(void* const* d_in, const int* in_sizes, int n_in,
                              void* d_out, int out_size, void* d_ws, size_t ws_size,
                              hipStream_t stream) {
    const float* x = (const float*)d_in[0];     // [2,2048,1024]
    const float* w = (const float*)d_in[1];     // [3072,1024]
    float* out = (float*)d_out;                 // [2,2048,1024]

    char* ws = (char*)d_ws;
    const size_t xb_bytes = (size_t)MTOT * DM * 2;                   // 8 MB
    const size_t wb_bytes = (size_t)NTOT * DM * 2;                   // 6 MB
    const size_t qb_bytes = (size_t)BATCH * HEADS * SEQ * HDIM * 2;  // 8 MB each

    unsigned short* Xb = (unsigned short*)(ws);
    unsigned short* Wb = (unsigned short*)(ws + xb_bytes);
    unsigned short* Qb = (unsigned short*)(ws + xb_bytes + wb_bytes);
    unsigned short* Kb = (unsigned short*)(ws + xb_bytes + wb_bytes + qb_bytes);
    unsigned short* Vb = (unsigned short*)(ws + xb_bytes + wb_bytes + 2 * qb_bytes);

    const int nx = MTOT * DM;       // 4194304
    const int nw = NTOT * DM;       // 3145728
    cvt_f32_to_bf16<<<nx / (256 * 4), 256, 0, stream>>>(x, Xb, nx);
    cvt_f32_to_bf16<<<nw / (256 * 4), 256, 0, stream>>>(w, Wb, nw);

    // 64 m-tiles x 48 n-tiles
    qkv_gemm<<<64 * 48, 128, 0, stream>>>(Xb, Wb, Qb, Kb, Vb);

    // 32 (b,h) pairs x 32 q-tiles
    attn_fwd<<<32 * 32, 128, 0, stream>>>(Qb, Kb, Vb, out);
}
